// DecoderLayer_3272765079831
// MI455X (gfx1250) — matrix operand order repaired and checked by reading, not compiled, not run on hardware
//
#include <hip/hip_runtime.h>
#include <hip/hip_bf16.h>
#include <math.h>

// ---------------------------------------------------------------------------
// CDNA5 (gfx1250) decoder layer.
//   * all matmuls on v_wmma_f32_16x16x32_bf16 (f32 accumulate)
//   * operands pre-converted to bf16 once (weights pre-packed in K-pairs)
//   * GEMM + attention tiles staged with GLOBAL_LOAD_ASYNC_TO_LDS_B128,
//     double-buffered, synchronized with s_wait_asynccnt (ASYNCcnt)
// ---------------------------------------------------------------------------

typedef __attribute__((ext_vector_type(16))) __bf16 v16bf;
typedef __attribute__((ext_vector_type(8)))  float  v8f;

union Frag16 { unsigned int u[8]; v16bf v; };

__device__ __forceinline__ unsigned int pack_bf16(float lo, float hi) {
  unsigned int l = (__float_as_uint(lo) + 0x8000u) >> 16;
  unsigned int h = (__float_as_uint(hi) + 0x8000u) >> 16;
  return (l & 0xffffu) | (h << 16);
}
__device__ __forceinline__ unsigned short f2bf(float x) {
  return (unsigned short)((__float_as_uint(x) + 0x8000u) >> 16);
}

// CDNA5 async global->LDS copy (16B per lane), tracked by ASYNCcnt.
__device__ __forceinline__ void async_b128(unsigned int lds_off, const void* gaddr) {
  asm volatile("global_load_async_to_lds_b128 %0, %1, off"
               :: "v"(lds_off), "v"(gaddr) : "memory");
}
__device__ __forceinline__ unsigned int lds_off_of(const void* p) {
  return (unsigned int)(unsigned long long)p;   // low 32 bits of AS3 address
}

// ---------------------------------------------------------------------------
// Elementwise f32 -> bf16 (8 elems/thread).
// ---------------------------------------------------------------------------
__global__ __launch_bounds__(256) void convert_bf16(
    const float* __restrict__ X, unsigned short* __restrict__ Y)
{
  size_t i = ((size_t)blockIdx.x * 256 + threadIdx.x) * 8;
  float4 a = *(const float4*)(X + i);
  float4 b = *(const float4*)(X + i + 4);
  uint4 o;
  o.x = pack_bf16(a.x, a.y); o.y = pack_bf16(a.z, a.w);
  o.z = pack_bf16(b.x, b.y); o.w = pack_bf16(b.z, b.w);
  *(uint4*)(Y + i) = o;
}

// ---------------------------------------------------------------------------
// Weight pre-pack: f32 [K,N] -> dword [K/2,N] holding bf16 pair (k, k+1).
// Grid: (N/256, K/2).
// ---------------------------------------------------------------------------
__global__ __launch_bounds__(256) void pack_weight(
    const float* __restrict__ W, unsigned int* __restrict__ Wp, int N)
{
  int n  = blockIdx.x * 256 + threadIdx.x;
  int kp = blockIdx.y;
  float lo = W[(size_t)(2 * kp)     * N + n];
  float hi = W[(size_t)(2 * kp + 1) * N + n];
  Wp[(size_t)kp * N + n] = pack_bf16(lo, hi);
}

// ---------------------------------------------------------------------------
// GEMM: C = A @ B.  A: bf16 [M,K] row-major.  B: pre-packed dword [K/2,N].
// EPI: 0 none, 1 +bias, 2 +bias+exact GELU.
// OUT: 0 f32 C;  1 bf16 Cb row-major;  2 bf16 Cb head-major [b,h,l,64]
//      (L=2048, 16 heads of 64: col = d*16+h).
// Block 256 (8 waves), tile 128x128, wave = 32x64, async double-buffered LDS.
// ---------------------------------------------------------------------------
template<int EPI, int OUT>
__global__ __launch_bounds__(256) void gemm_bf16_wmma(
    const unsigned short* __restrict__ A, const unsigned int* __restrict__ Bp,
    const float* __restrict__ bias,
    float* __restrict__ C, unsigned short* __restrict__ Cb,
    int M, int N, int K)
{
  __shared__ __align__(16) unsigned int AsU[2][128 * 20]; // 128 x 16 dwords, pitch 20 (80B)
  __shared__ __align__(16) unsigned int BpU[2][16 * 132]; // 16 x 128 dwords, pitch 132 (528B)

  const int tid  = threadIdx.x;
  const int lane = tid & 31;
  const int wid  = tid >> 5;
  const int half = lane >> 4;
  const int lj   = lane & 15;
  const int wr   = wid & 3;
  const int wc   = wid >> 2;
  const int rowBase = blockIdx.y * 128;
  const int colBase = blockIdx.x * 128;

  v8f acc[2][4];
#pragma unroll
  for (int i = 0; i < 2; ++i)
#pragma unroll
    for (int j = 0; j < 4; ++j)
      acc[i][j] = (v8f){0.f,0.f,0.f,0.f,0.f,0.f,0.f,0.f};

  // async stage of one (A,B) tile pair into buffer `buf` (4 issues/thread)
  auto stage = [&](int buf, int kk) {
    unsigned aB = lds_off_of(&AsU[buf][0]);
    unsigned bB = lds_off_of(&BpU[buf][0]);
#pragma unroll
    for (int s = 0; s < 2; ++s) {
      int id = tid + s * 256;        // 0..511
      int r = id >> 2, seg = id & 3; // row, 16B segment of 64B row
      async_b128(aB + (unsigned)(r * 20 + seg * 4) * 4,
                 A + (size_t)(rowBase + r) * K + kk + seg * 8);
    }
#pragma unroll
    for (int s = 0; s < 2; ++s) {
      int id = tid + s * 256;
      int rp = id >> 5, seg = id & 31;             // k-pair row, 16B segment
      int col = (seg * 4) ^ ((rp >> 3) << 4);      // half-XOR bank swizzle
      async_b128(bB + (unsigned)(rp * 132 + col) * 4,
                 Bp + (size_t)((kk >> 1) + rp) * N + colBase + seg * 4);
    }
  };

  stage(0, 0);
  int cur = 0;
  for (int kk = 0; kk < K; kk += 32) {
    const bool last = (kk + 32 >= K);
    if (!last) stage(cur ^ 1, kk + 32);                 // prefetch next tile
    if (last) asm volatile("s_wait_asynccnt 0" ::: "memory");
    else      asm volatile("s_wait_asynccnt 4" ::: "memory"); // keep prefetch inflight
    __syncthreads();

    Frag16 af[2], bfg[4];
#pragma unroll
    for (int mt = 0; mt < 2; ++mt) {
      int r = wr * 32 + mt * 16 + lj;
#pragma unroll
      for (int v = 0; v < 8; ++v) {
        int ku = half * 8 + v;
        af[mt].u[v] = AsU[cur][r * 20 + ku];
      }
    }
#pragma unroll
    for (int nt = 0; nt < 4; ++nt) {
      int c = (wc * 64 + nt * 16 + lj) ^ (half << 4);
#pragma unroll
      for (int v = 0; v < 8; ++v)
        bfg[nt].u[v] = BpU[cur][(half * 8 + v) * 132 + c];
    }
#pragma unroll
    for (int mt = 0; mt < 2; ++mt)
#pragma unroll
      for (int nt = 0; nt < 4; ++nt)
        acc[mt][nt] = __builtin_amdgcn_wmma_f32_16x16x32_bf16(
            false, af[mt].v, false, bfg[nt].v, (short)0, acc[mt][nt], false, false);
    __syncthreads();
    cur ^= 1;
  }

#pragma unroll
  for (int mt = 0; mt < 2; ++mt)
#pragma unroll
    for (int nt = 0; nt < 4; ++nt) {
      int col = colBase + wc * 64 + nt * 16 + lj;
      float bv = (EPI > 0) ? bias[col] : 0.f;
#pragma unroll
      for (int v = 0; v < 8; ++v) {
        int row = rowBase + wr * 32 + mt * 16 + half * 8 + v;
        float xv = acc[mt][nt][v];
        if (EPI > 0) xv += bv;
        if (EPI == 2) xv = 0.5f * xv * (1.f + erff(xv * 0.70710678118654752f));
        if (OUT == 0) {
          C[(size_t)row * N + col] = xv;
        } else if (OUT == 1) {
          Cb[(size_t)row * N + col] = f2bf(xv);
        } else {  // head-major bf16: [b, h, l, d]
          int bb = row >> 11, l = row & 2047, h = col & 15, d = col >> 4;
          Cb[(((size_t)((bb << 4) + h)) * 2048 + l) * 64 + d] = f2bf(xv);
        }
      }
    }
}

// ---------------------------------------------------------------------------
// Attention (faithful to source): S = Q K^T per head, mask BEFORE 1/8 scale,
// online softmax over m, context uses K as values.
// Q/K in head-major bf16 [B*16, L, 64]; ctx out row-major bf16 (col=d*16+h).
// Grid (L/128, 16, B); block 256 = 8 waves; wave = 16 query rows.
// ---------------------------------------------------------------------------
template<bool HASMASK>
__global__ __launch_bounds__(256) void attn_wmma(
    const unsigned short* __restrict__ Qb, const unsigned short* __restrict__ Kb,
    const unsigned char* __restrict__ mask, unsigned short* __restrict__ Outb,
    int L, int M)
{
  __shared__ __align__(16) unsigned int KtU[2][32 * 36];   // 32 m x 32 dwords, pitch 36
  __shared__ __align__(4) unsigned short PbS[8][16 * 34];  // per-wave P transpose
  __shared__ unsigned char maskS[2][32];

  const int tid  = threadIdx.x;
  const int lane = tid & 31;
  const int wid  = tid >> 5;
  const int half = lane >> 4;
  const int lj   = lane & 15;
  const int head = blockIdx.y;
  const int b    = blockIdx.z;
  const int bh   = b * 16 + head;
  const int q0   = blockIdx.x * 128 + wid * 16;

  const unsigned short* Kbase = Kb + (size_t)bh * M * 64;

  // Q A-fragments: contiguous head-major rows -> four 16B vector loads.
  Frag16 qa[2];
  {
    const uint4* q4 = (const uint4*)(Qb + ((size_t)bh * L + q0 + lj) * 64);
#pragma unroll
    for (int c = 0; c < 2; ++c) {
      uint4 lo = q4[c * 4 + 2 * half];
      uint4 hi = q4[c * 4 + 2 * half + 1];
      qa[c].u[0] = lo.x; qa[c].u[1] = lo.y; qa[c].u[2] = lo.z; qa[c].u[3] = lo.w;
      qa[c].u[4] = hi.x; qa[c].u[5] = hi.y; qa[c].u[6] = hi.z; qa[c].u[7] = hi.w;
    }
  }

  v8f o[4];
#pragma unroll
  for (int dt = 0; dt < 4; ++dt)
    o[dt] = (v8f){0.f,0.f,0.f,0.f,0.f,0.f,0.f,0.f};
  float rm[8], rl[8];
#pragma unroll
  for (int v = 0; v < 8; ++v) { rm[v] = -1e30f; rl[v] = 0.f; }

  auto stageK = [&](int buf, int m0) {
    unsigned kB = lds_off_of(&KtU[buf][0]);
    int m = tid >> 3, seg = tid & 7;                 // 32 rows x 8 segments (8 segs of 16B)
    async_b128(kB + (unsigned)(m * 36 + seg * 4) * 4,
               Kbase + (size_t)(m0 + m) * 64 + seg * 8);
    if (HASMASK && tid < 32) maskS[buf][tid] = mask[(size_t)b * M + m0 + tid];
  };

  stageK(0, 0);
  int cur = 0;
  for (int m0 = 0; m0 < M; m0 += 32) {
    const bool last = (m0 + 32 >= M);
    if (!last) stageK(cur ^ 1, m0 + 32);
    if (last) asm volatile("s_wait_asynccnt 0" ::: "memory");
    else      asm volatile("s_wait_asynccnt 1" ::: "memory");
    __syncthreads();

    // Scores for two 16-wide m tiles; S column = lane lj.
    float p[2][8];
#pragma unroll
    for (int mt = 0; mt < 2; ++mt) {
      Frag16 kb0, kb1;
      int mrow = mt * 16 + lj;
#pragma unroll
      for (int v = 0; v < 8; ++v) {
        kb0.u[v] = KtU[cur][mrow * 36 + (half * 8 + v)];       // d 0..31
        kb1.u[v] = KtU[cur][mrow * 36 + 16 + (half * 8 + v)];  // d 32..63
      }
      v8f s = (v8f){0.f,0.f,0.f,0.f,0.f,0.f,0.f,0.f};
      s = __builtin_amdgcn_wmma_f32_16x16x32_bf16(false, qa[0].v, false, kb0.v, (short)0, s, false, false);
      s = __builtin_amdgcn_wmma_f32_16x16x32_bf16(false, qa[1].v, false, kb1.v, (short)0, s, false, false);
      bool mv = HASMASK ? (maskS[cur][mrow] != 0) : false;
#pragma unroll
      for (int v = 0; v < 8; ++v) {
        float xv = s[v];
        if (HASMASK && mv) xv = -1e9f;   // mask BEFORE scale (source order)
        p[mt][v] = xv * 0.125f;
      }
    }

    // Online softmax per row (row = half*8+v across 16 lanes of the half).
#pragma unroll
    for (int v = 0; v < 8; ++v) {
      float mx = fmaxf(p[0][v], p[1][v]);
#pragma unroll
      for (int off = 1; off < 16; off <<= 1)
        mx = fmaxf(mx, __shfl_xor(mx, off, 32));
      float nm   = fmaxf(rm[v], mx);
      float corr = __expf(rm[v] - nm);
      rm[v] = nm;
      float e0 = __expf(p[0][v] - nm);
      float e1 = __expf(p[1][v] - nm);
      p[0][v] = e0; p[1][v] = e1;
      float rs = e0 + e1;
#pragma unroll
      for (int off = 1; off < 16; off <<= 1)
        rs += __shfl_xor(rs, off, 32);
      rl[v] = rl[v] * corr + rs;
#pragma unroll
      for (int dt = 0; dt < 4; ++dt) o[dt][v] *= corr;
    }

    // Transpose P (C-layout) -> A-layout through per-wave LDS buffer.
#pragma unroll
    for (int v = 0; v < 8; ++v) {
      int row = half * 8 + v;
      PbS[wid][row * 34 + lj]      = f2bf(p[0][v]);
      PbS[wid][row * 34 + 16 + lj] = f2bf(p[1][v]);
    }
    Frag16 pf;
    {
      const unsigned int* pb = (const unsigned int*)&PbS[wid][0];
#pragma unroll
      for (int v = 0; v < 8; ++v) {
        int k0 = half * 16 + 2 * v;
        pf.u[v] = pb[(lj * 34 + k0) >> 1];
      }
    }

    // P @ K (values = K per source). B frag: contraction m, N = d.
#pragma unroll
    for (int dt = 0; dt < 4; ++dt) {
      Frag16 kb;
      int dcol = dt * 16 + lj;
#pragma unroll
      for (int v = 0; v < 8; ++v) {
        int m = half * 16 + 2 * v;
        unsigned int a0 = KtU[cur][m * 36 + (dcol >> 1)];
        unsigned int a1 = KtU[cur][(m + 1) * 36 + (dcol >> 1)];
        unsigned int lo = (dcol & 1) ? (a0 >> 16) : (a0 & 0xffffu);
        unsigned int hi = (dcol & 1) ? (a1 >> 16) : (a1 & 0xffffu);
        kb.u[v] = lo | (hi << 16);
      }
      o[dt] = __builtin_amdgcn_wmma_f32_16x16x32_bf16(false, pf.v, false, kb.v, (short)0, o[dt], false, false);
    }
    __syncthreads();
    cur ^= 1;
  }

  // Normalize and store row-major bf16: out[b*L + l, d*16 + head].
#pragma unroll
  for (int dt = 0; dt < 4; ++dt)
#pragma unroll
    for (int v = 0; v < 8; ++v) {
      float inv = 1.f / rl[v];
      int row = q0 + half * 8 + v;
      Outb[(size_t)(b * L + row) * 1024 + (dt * 16 + lj) * 16 + head] = f2bf(o[dt][v] * inv);
    }
}

// ---------------------------------------------------------------------------
// Y = LayerNorm(X + R) * g + b   (D=1024, biased var, eps=1e-5).
// Optionally also writes bf16 copy Yb (input to the next GEMM).
// In-place safe for Y == X (each thread reads its own 4 elems before writing).
// ---------------------------------------------------------------------------
template<bool WBF>
__global__ __launch_bounds__(256) void add_layernorm(
    const float* __restrict__ X, const float* __restrict__ R,
    const float* __restrict__ g, const float* __restrict__ bta,
    float* __restrict__ Y, unsigned short* __restrict__ Yb)
{
  const int row = blockIdx.x;
  const int tid = threadIdx.x;
  const size_t base = (size_t)row * 1024 + tid * 4;
  float4 xv = *(const float4*)(X + base);
  float4 rv = *(const float4*)(R + base);
  float a0 = xv.x + rv.x, a1 = xv.y + rv.y, a2 = xv.z + rv.z, a3 = xv.w + rv.w;
  float s = a0 + a1 + a2 + a3;
  float q = a0 * a0 + a1 * a1 + a2 * a2 + a3 * a3;
#pragma unroll
  for (int off = 16; off >= 1; off >>= 1) {
    s += __shfl_xor(s, off, 32);
    q += __shfl_xor(q, off, 32);
  }
  __shared__ float reds[8], redq[8];
  const int wid = tid >> 5, lane = tid & 31;
  if (lane == 0) { reds[wid] = s; redq[wid] = q; }
  __syncthreads();
  float ts = 0.f, tq = 0.f;
#pragma unroll
  for (int i = 0; i < 8; ++i) { ts += reds[i]; tq += redq[i]; }
  const float mean = ts * (1.f / 1024.f);
  const float var  = tq * (1.f / 1024.f) - mean * mean;
  const float rstd = rsqrtf(var + 1e-5f);
  float4 gv = *(const float4*)(g + tid * 4);
  float4 bv = *(const float4*)(bta + tid * 4);
  float y0 = (a0 - mean) * rstd * gv.x + bv.x;
  float y1 = (a1 - mean) * rstd * gv.y + bv.y;
  float y2 = (a2 - mean) * rstd * gv.z + bv.z;
  float y3 = (a3 - mean) * rstd * gv.w + bv.w;
  float4 yv; yv.x = y0; yv.y = y1; yv.z = y2; yv.w = y3;
  *(float4*)(Y + base) = yv;
  if (WBF) {
    uint2 o; o.x = pack_bf16(y0, y1); o.y = pack_bf16(y2, y3);
    *(uint2*)(Yb + base) = o;
  }
}

// ---------------------------------------------------------------------------
// Host-side pipeline.
// ---------------------------------------------------------------------------
extern "C" void kernel_launch(void* const* d_in, const int* in_sizes, int n_in,
                              void* d_out, int out_size, void* d_ws, size_t ws_size,
                              hipStream_t stream)
{
  (void)in_sizes; (void)n_in; (void)out_size; (void)ws_size;

  const float* x    = (const float*)d_in[0];
  const float* enc  = (const float*)d_in[1];
  const unsigned char* mask = (const unsigned char*)d_in[2];
  const float* q1W  = (const float*)d_in[3];
  const float* w1W  = (const float*)d_in[4];
  const float* o1W  = (const float*)d_in[5];
  const float* q2W  = (const float*)d_in[6];
  const float* w2W  = (const float*)d_in[7];
  const float* o2W  = (const float*)d_in[8];
  const float* ffW1 = (const float*)d_in[9];
  const float* ffb1 = (const float*)d_in[10];
  const float* ffW2 = (const float*)d_in[11];
  const float* ffb2 = (const float*)d_in[12];
  const float* g1   = (const float*)d_in[13];
  const float* b1   = (const float*)d_in[14];
  const float* g2   = (const float*)d_in[15];
  const float* b2   = (const float*)d_in[16];
  const float* g3   = (const float*)d_in[17];
  const float* b3   = (const float*)d_in[18];

  char* ws = (char*)d_ws;
  const size_t MB = 1024 * 1024;
  float*          bufA   = (float*)         (ws + 0 * MB);   // 16 MB f32 scratch
  unsigned short* qbf    = (unsigned short*)(ws + 16 * MB);  // [B*16, L, 64]
  unsigned short* kbf    = (unsigned short*)(ws + 24 * MB);
  unsigned short* ctxbf  = (unsigned short*)(ws + 32 * MB);  // [4096,1024] bf16
  unsigned short* xbf    = (unsigned short*)(ws + 40 * MB);
  unsigned short* encbf  = (unsigned short*)(ws + 48 * MB);
  unsigned short* xcurbf = (unsigned short*)(ws + 56 * MB);  // x1bf then x2bf
  unsigned short* hbf    = (unsigned short*)(ws + 64 * MB);  // 32 MB FFN hidden
  unsigned int*   wq1    = (unsigned int*)  (ws + 96 * MB);
  unsigned int*   ww1    = (unsigned int*)  (ws + 98 * MB);
  unsigned int*   wo1    = (unsigned int*)  (ws + 100 * MB);
  unsigned int*   wq2    = (unsigned int*)  (ws + 102 * MB);
  unsigned int*   ww2    = (unsigned int*)  (ws + 104 * MB);
  unsigned int*   wo2    = (unsigned int*)  (ws + 106 * MB);
  unsigned int*   wff1   = (unsigned int*)  (ws + 108 * MB); // 8 MB
  unsigned int*   wff2   = (unsigned int*)  (ws + 116 * MB); // 8 MB
  float* xout = (float*)d_out;  // x1 (f32) -> x2 in place -> final output

  const int Mrows = 4096;  // B*L
  dim3 blk(256);
  dim3 gP (1024 / 128, Mrows / 128);
  dim3 gF1(4096 / 128, Mrows / 128);
  dim3 gAttn(2048 / 128, 16, 2);

  // ---- one-time bf16 conversion / packing ----
  convert_bf16<<<2048, blk, 0, stream>>>(x,   xbf);
  convert_bf16<<<2048, blk, 0, stream>>>(enc, encbf);
  pack_weight<<<dim3(4, 512),  blk, 0, stream>>>(q1W,  wq1,  1024);
  pack_weight<<<dim3(4, 512),  blk, 0, stream>>>(w1W,  ww1,  1024);
  pack_weight<<<dim3(4, 512),  blk, 0, stream>>>(o1W,  wo1,  1024);
  pack_weight<<<dim3(4, 512),  blk, 0, stream>>>(q2W,  wq2,  1024);
  pack_weight<<<dim3(4, 512),  blk, 0, stream>>>(w2W,  ww2,  1024);
  pack_weight<<<dim3(4, 512),  blk, 0, stream>>>(o2W,  wo2,  1024);
  pack_weight<<<dim3(16, 512), blk, 0, stream>>>(ffW1, wff1, 4096);
  pack_weight<<<dim3(4, 2048), blk, 0, stream>>>(ffW2, wff2, 1024);

  // ---- self-attention ----
  gemm_bf16_wmma<0,2><<<gP, blk, 0, stream>>>(xbf, wq1, nullptr, nullptr, qbf, Mrows, 1024, 1024);
  gemm_bf16_wmma<0,2><<<gP, blk, 0, stream>>>(xbf, ww1, nullptr, nullptr, kbf, Mrows, 1024, 1024);
  attn_wmma<false><<<gAttn, blk, 0, stream>>>(qbf, kbf, nullptr, ctxbf, 2048, 2048);
  gemm_bf16_wmma<0,0><<<gP, blk, 0, stream>>>(ctxbf, wo1, nullptr, bufA, nullptr, Mrows, 1024, 1024);
  add_layernorm<true><<<Mrows, blk, 0, stream>>>(x, bufA, g1, b1, xout, xcurbf);

  // ---- cross-attention (masked) ----
  gemm_bf16_wmma<0,2><<<gP, blk, 0, stream>>>(xcurbf, wq2, nullptr, nullptr, qbf, Mrows, 1024, 1024);
  gemm_bf16_wmma<0,2><<<gP, blk, 0, stream>>>(encbf,  ww2, nullptr, nullptr, kbf, Mrows, 1024, 1024);
  attn_wmma<true><<<gAttn, blk, 0, stream>>>(qbf, kbf, mask, ctxbf, 2048, 2048);
  gemm_bf16_wmma<0,0><<<gP, blk, 0, stream>>>(ctxbf, wo2, nullptr, bufA, nullptr, Mrows, 1024, 1024);
  add_layernorm<true><<<Mrows, blk, 0, stream>>>(xout, bufA, g2, b2, xout, xcurbf);  // x2 in place

  // ---- feed-forward ----
  gemm_bf16_wmma<2,1><<<gF1, blk, 0, stream>>>(xcurbf, wff1, ffb1, nullptr, hbf, Mrows, 4096, 1024);
  gemm_bf16_wmma<1,0><<<gP,  blk, 0, stream>>>(hbf,    wff2, ffb2, bufA, nullptr, Mrows, 1024, 4096);
  add_layernorm<false><<<Mrows, blk, 0, stream>>>(xout, bufA, g3, b3, xout, nullptr);
}
